// interAUGraphConv_29025388986856
// MI455X (gfx1250) — matrix operand order repaired and checked by reading, not compiled, not run on hardware
//
#include <hip/hip_runtime.h>

// ---------------------------------------------------------------------------
// interAUGraphConv for MI455X (gfx1250, wave32, WMMA)
//   pre = (adj @ (w/17)) @ inputs   -- two GEMMs via v_wmma_f32_16x16x32_bf16
//   bf16x3 split precision: x = hi + lo; x*y ~= hi*hi + hi*lo + lo*hi  (f32 acc)
//   B matrices pre-transposed in global ws so A and B^T tiles stage as
//   contiguous b128 copies -> async global->LDS (ASYNCcnt) double buffering.
//   Stage-2 epilogue fused: out = ((0.6<pre<=1)?pre:1)/prob[row]; mask=pre>0.6
//   Output stored non-temporal (512MB streaming) to keep operands in L2.
// ---------------------------------------------------------------------------

typedef __attribute__((ext_vector_type(16))) __bf16 v16bf;
typedef __attribute__((ext_vector_type(8)))  float  v8f;
typedef __attribute__((ext_vector_type(4)))  int    v4i;
typedef unsigned short ushort_t;

union Frag { v16bf v; uint4 u[2]; };

#if defined(__gfx1250__) && __has_builtin(__builtin_amdgcn_global_load_async_to_lds_b128)
#define HAVE_ASYNC_LDS 1
#endif

__device__ __forceinline__ void cp_b128(const ushort_t* g, ushort_t* l) {
#ifdef HAVE_ASYNC_LDS
  ushort_t* gnc = (ushort_t*)g;  // drop const, then retype + addrspace cast
  __builtin_amdgcn_global_load_async_to_lds_b128(
      (__attribute__((address_space(1))) v4i*)gnc,
      (__attribute__((address_space(3))) v4i*)l, 0, 0);
#else
  *(uint4*)l = *(const uint4*)g;
#endif
}

__device__ __forceinline__ void wait_async_all() {
#ifdef HAVE_ASYNC_LDS
#if __has_builtin(__builtin_amdgcn_s_wait_asynccnt)
  __builtin_amdgcn_s_wait_asynccnt(0);
#else
  asm volatile("s_wait_asynccnt 0" ::: "memory");
#endif
#endif
}

__device__ __forceinline__ ushort_t f2bf(float x) {
  union { float f; unsigned u; } a; a.f = x;
  unsigned u = a.u;
  u += 0x7FFFu + ((u >> 16) & 1u);          // round-to-nearest-even to bf16
  return (ushort_t)(u >> 16);
}
__device__ __forceinline__ float bf2f(ushort_t h) {
  union { float f; unsigned u; } a; a.u = ((unsigned)h) << 16; return a.f;
}

// f32 -> (bf16 hi, bf16 lo) split with optional scale (A-side matrices)
__global__ __launch_bounds__(256)
void split_bf16x2(const float* __restrict__ x, ushort_t* __restrict__ hi,
                  ushort_t* __restrict__ lo, float scale, int n) {
  for (int i = blockIdx.x * blockDim.x + threadIdx.x; i < n; i += gridDim.x * blockDim.x) {
    float v = x[i] * scale;
    ushort_t h = f2bf(v);
    hi[i] = h;
    lo[i] = f2bf(v - bf2f(h));
  }
}

// f32 [R][C] -> transposed bf16 hi/lo [C][R], LDS-tiled 32x32 (B-side matrices)
__global__ __launch_bounds__(256)
void split_bf16x2_T(const float* __restrict__ x, ushort_t* __restrict__ hiT,
                    ushort_t* __restrict__ loT, float scale, int R, int C) {
  __shared__ ushort_t th[32][33];
  __shared__ ushort_t tl[32][33];
  const int r0 = blockIdx.y * 32, c0 = blockIdx.x * 32;
  const int tx = threadIdx.x & 31, ty = threadIdx.x >> 5;
#pragma unroll
  for (int i = ty; i < 32; i += 8) {
    float v = x[(size_t)(r0 + i) * C + c0 + tx] * scale;
    ushort_t h = f2bf(v);
    th[i][tx] = h;
    tl[i][tx] = f2bf(v - bf2f(h));
  }
  __syncthreads();
#pragma unroll
  for (int i = ty; i < 32; i += 8) {
    hiT[(size_t)(c0 + i) * R + r0 + tx] = th[tx][i];
    loT[(size_t)(c0 + i) * R + r0 + tx] = tl[tx][i];
  }
}

// ---------------------------------------------------------------------------
// GEMM: C[?,Ncols] = A[?,K] * B[K,Ncols], with B supplied TRANSPOSED (row
// stride K). 256 thr = 8 waves; 128x128 C tile; wave grid 2(M) x 4(N); each
// wave 4x2 WMMA tiles. K staged 32 wide, double-buffered async global->LDS.
// MODE 0: write f32 C.   MODE 1: fused activation epilogue (NT stores).
// ---------------------------------------------------------------------------
template <int MODE>
__global__ __launch_bounds__(256)
void gemm_bf16x3(const ushort_t* __restrict__ Ahi, const ushort_t* __restrict__ Alo,
                 const ushort_t* __restrict__ Bthi, const ushort_t* __restrict__ Btlo,
                 int Ncols, int K,
                 float* __restrict__ Cout,
                 float* __restrict__ Out, float* __restrict__ Mask,
                 const float* __restrict__ Prob) {
  // [buf][hi=0/lo=1][128 rows x 32 k]  : 8 x 8KB = 64KB
  __shared__ __align__(16) ushort_t sA[2][2][128 * 32];
  __shared__ __align__(16) ushort_t sB[2][2][128 * 32];

  const int tid  = threadIdx.x;
  const int lane = tid & 31;
  const int wv   = tid >> 5;
  const int wm   = wv & 1;        // wave M group: 0..1  (64 rows each)
  const int wn   = wv >> 1;       // wave N group: 0..3  (32 cols each)
  const int mb   = blockIdx.y * 128;
  const int nb   = blockIdx.x * 128;

  const int mrow = lane & 15;     // M (A frag) / N (B frag) within 16x16 tile
  const int half = lane >> 4;     // K-block selector per ISA 7.12.2 layout

  // stage one 128x32 bf16 tile pair set into LDS buffer `buf`
  auto fill = [&](int buf, int kt) {
#pragma unroll
    for (int it = 0; it < 2; ++it) {
      const int c = tid + it * 256;         // 512 chunks of 16B per tile
      const int r = c >> 2;
      const int o = (c & 3) << 3;           // ushort offset (8 ushort = 16B)
      const size_t ga = (size_t)(mb + r) * K + kt + o;
      const size_t gb = (size_t)(nb + r) * K + kt + o;
      const int lofs = r * 32 + o;
      cp_b128(Ahi  + ga, &sA[buf][0][lofs]);
      cp_b128(Alo  + ga, &sA[buf][1][lofs]);
      cp_b128(Bthi + gb, &sB[buf][0][lofs]);
      cp_b128(Btlo + gb, &sB[buf][1][lofs]);
    }
  };

  v8f acc[4][2];
#pragma unroll
  for (int mt = 0; mt < 4; ++mt)
#pragma unroll
    for (int nt = 0; nt < 2; ++nt)
#pragma unroll
      for (int e = 0; e < 8; ++e) acc[mt][nt][e] = 0.0f;

  fill(0, 0);
  wait_async_all();
  __syncthreads();

  int cur = 0;
  for (int kt = 0; kt < K; kt += 32) {
    if (kt + 32 < K) fill(cur ^ 1, kt + 32);   // overlap with compute below

    // --- fragments from LDS (layouts per CDNA5 ISA 7.12.2) ---
    Frag ahi[4], alo[4], bhi[2], blo[2];
#pragma unroll
    for (int mt = 0; mt < 4; ++mt) {
      const int r = wm * 64 + mt * 16 + mrow;
      const ushort_t* ph = &sA[cur][0][r * 32 + half * 8];
      const ushort_t* pl = &sA[cur][1][r * 32 + half * 8];
      ahi[mt].u[0] = *(const uint4*)(ph);        // K = half*8 .. +7   (v0-3)
      ahi[mt].u[1] = *(const uint4*)(ph + 16);   // K = 16+half*8 ..   (v4-7)
      alo[mt].u[0] = *(const uint4*)(pl);
      alo[mt].u[1] = *(const uint4*)(pl + 16);
    }
#pragma unroll
    for (int nt = 0; nt < 2; ++nt) {
      const int n = wn * 32 + nt * 16 + mrow;
      const ushort_t* ph = &sB[cur][0][n * 32 + half * 8];
      const ushort_t* pl = &sB[cur][1][n * 32 + half * 8];
      bhi[nt].u[0] = *(const uint4*)(ph);        // K = half*8 .. +7
      bhi[nt].u[1] = *(const uint4*)(ph + 16);   // K = 16+half*8 .. +7
      blo[nt].u[0] = *(const uint4*)(pl);
      blo[nt].u[1] = *(const uint4*)(pl + 16);
    }

    // --- bf16x3: hi*hi + hi*lo + lo*hi, one f32 accumulator ---
#pragma unroll
    for (int mt = 0; mt < 4; ++mt)
#pragma unroll
      for (int nt = 0; nt < 2; ++nt) {
        acc[mt][nt] = __builtin_amdgcn_wmma_f32_16x16x32_bf16(
            false, ahi[mt].v, false, bhi[nt].v, (short)0, acc[mt][nt], false, false);
        acc[mt][nt] = __builtin_amdgcn_wmma_f32_16x16x32_bf16(
            false, ahi[mt].v, false, blo[nt].v, (short)0, acc[mt][nt], false, false);
        acc[mt][nt] = __builtin_amdgcn_wmma_f32_16x16x32_bf16(
            false, alo[mt].v, false, bhi[nt].v, (short)0, acc[mt][nt], false, false);
      }

    wait_async_all();   // next buffer filled (this wave's async ops)
    __syncthreads();    // all waves: done reading cur, next buffer visible
    cur ^= 1;
  }

  // --- epilogue: C/D layout -> M = v + 8*half, N = lane&15 ---
  const int ncol = lane & 15;
#pragma unroll
  for (int mt = 0; mt < 4; ++mt) {
    const int rbase = mb + wm * 64 + mt * 16 + 8 * half;
#pragma unroll
    for (int nt = 0; nt < 2; ++nt) {
      const int c = nb + wn * 32 + nt * 16 + ncol;
#pragma unroll
      for (int v = 0; v < 8; ++v) {
        const int r = rbase + v;
        const size_t idx = (size_t)r * Ncols + c;
        const float pre = acc[mt][nt][v];
        if (MODE == 0) {
          Cout[idx] = pre;                         // re-read right away: keep RT
        } else {
          const float act = (pre > 0.6f && pre <= 1.0f) ? pre : 1.0f;
          __builtin_nontemporal_store(act / Prob[r], &Out[idx]);
          __builtin_nontemporal_store((pre > 0.6f) ? 1.0f : 0.0f, &Mask[idx]);
        }
      }
    }
  }
}

// ---------------------------------------------------------------------------
extern "C" void kernel_launch(void* const* d_in, const int* in_sizes, int n_in,
                              void* d_out, int out_size, void* d_ws, size_t ws_size,
                              hipStream_t stream) {
  (void)in_sizes; (void)n_in; (void)out_size; (void)ws_size;

  constexpr int N = 8192, IN = 1024, OUT = 1024, D = 8192;

  const float* inputs = (const float*)d_in[0];   // [OUT, D]
  const float* adj    = (const float*)d_in[1];   // [N, IN]
  const float* prob   = (const float*)d_in[2];   // [N]
  const float* w      = (const float*)d_in[3];   // [IN, OUT]

  float* out  = (float*)d_out;                   // [N, D]
  float* mask = (float*)d_out + (size_t)N * D;   // [N, D] (bool as 0/1 f32)

  // workspace carve-up (byte offsets)
  char* ws = (char*)d_ws;
  ushort_t* adjhi = (ushort_t*)(ws + ((size_t)0   << 20));  // 16 MB [N][IN]
  ushort_t* adjlo = (ushort_t*)(ws + ((size_t)16  << 20));  // 16 MB
  ushort_t* wThi  = (ushort_t*)(ws + ((size_t)32  << 20));  //  2 MB [OUT][IN]
  ushort_t* wTlo  = (ushort_t*)(ws + ((size_t)34  << 20));  //  2 MB
  ushort_t* inThi = (ushort_t*)(ws + ((size_t)36  << 20));  // 16 MB [D][OUT]
  ushort_t* inTlo = (ushort_t*)(ws + ((size_t)52  << 20));  // 16 MB
  float*    tmp   = (float*)   (ws + ((size_t)68  << 20));  // 32 MB [N][OUT]
  ushort_t* thi   = (ushort_t*)(ws + ((size_t)100 << 20));  // 16 MB [N][OUT]
  ushort_t* tlo   = (ushort_t*)(ws + ((size_t)116 << 20));  // 16 MB

  // 1) split A-side matrices; split+transpose B-side matrices (w gets /17)
  {
    int n1 = N * IN;
    split_bf16x2<<<dim3((n1 + 255) / 256), dim3(256), 0, stream>>>(
        adj, adjhi, adjlo, 1.0f, n1);
    split_bf16x2_T<<<dim3(OUT / 32, IN / 32), dim3(256), 0, stream>>>(
        w, wThi, wTlo, 1.0f / 17.0f, IN, OUT);
    split_bf16x2_T<<<dim3(D / 32, OUT / 32), dim3(256), 0, stream>>>(
        inputs, inThi, inTlo, 1.0f, OUT, D);
  }

  // 2) tmp[N,OUT] = adj @ (w/17)
  gemm_bf16x3<0><<<dim3(OUT / 128, N / 128), dim3(256), 0, stream>>>(
      adjhi, adjlo, wThi, wTlo, OUT, IN, tmp, nullptr, nullptr, nullptr);

  // 3) split tmp
  {
    int n4 = N * OUT;
    split_bf16x2<<<dim3((n4 + 255) / 256), dim3(256), 0, stream>>>(
        tmp, thi, tlo, 1.0f, n4);
  }

  // 4) pre = tmp @ inputs, fused activation + /prob + mask
  gemm_bf16x3<1><<<dim3(D / 128, N / 128), dim3(256), 0, stream>>>(
      thi, tlo, inThi, inTlo, D, OUT, nullptr, out, mask, prob);
}
